// RNNGenerator_48979807044187
// MI455X (gfx1250) — matrix operand order repaired and checked by reading, not compiled, not run on hardware
//
#include <hip/hip_runtime.h>
#include <math.h>

typedef __attribute__((ext_vector_type(16))) _Float16 v16h;
typedef __attribute__((ext_vector_type(8)))  _Float16 v8h;
typedef __attribute__((ext_vector_type(8)))  float    v8f;
typedef __attribute__((ext_vector_type(4)))  float    v4f;

#define HID   128
#define BATCH 64
#define SEQ   2048
#define RB    (SEQ * BATCH)          // rows of the projection GEMM (131072)
#define HPAD  (HID + 8)              // padded h row (kills ds bank conflicts)

// LDS layout carved from dynamic shared memory:
//   pbuf: 2 x [BATCH*HID] f32   (double-buffered async-prefetched P[s] tile)
//   hbuf: 2 x [BATCH][HPAD] f16 (double-buffered hidden state)
#define PBUF_BYTES (2 * BATCH * HID * 4)
#define HBUF_BYTES (2 * BATCH * HPAD * 2)
#define SMEM_BYTES (PBUF_BYTES + HBUF_BYTES)

// generic -> LDS(addrspace 3) byte offset for async-to-LDS instructions
__device__ __forceinline__ unsigned lds_off(const void* p) {
    return (unsigned)(size_t)(__attribute__((address_space(3))) const char*)p;
}

// Build a WMMA B fragment (16x16x32 f16) for K-chunk kc, N-tile nt from an
// f32 weight matrix W[HID][HID]; GEMM needs B(k,n) = W[n][k] (contiguous in k).
// Layout (ISA 7.12.2): lanes 0-15 hold K=kb..kb+7,kb+16..kb+23 of column n,
// lanes 16-31 hold K=kb+8..kb+15,kb+24..kb+31.
__device__ __forceinline__ v16h load_w_frag(const float* __restrict__ W,
                                            int kc, int nt, int lane) {
    int col = nt * 16 + (lane & 15);
    int kb  = kc * 32 + ((lane >> 4) << 3);
    const v4f* p = (const v4f*)(W + col * HID + kb);
    v4f w0 = p[0], w1 = p[1], w2 = p[4], w3 = p[5];
    v16h b;
#pragma unroll
    for (int i = 0; i < 4; ++i) {
        b[i]      = (_Float16)w0[i];
        b[i + 4]  = (_Float16)w1[i];
        b[i + 8]  = (_Float16)w2[i];
        b[i + 12] = (_Float16)w3[i];
    }
    return b;
}

// ---------------------------------------------------------------------------
// Projection GEMM: P[r][h] = A[r][:] . Wih[h][:] + bih[h] + bhh[h]
// rows r = s*BATCH + b.  A element (r,k) at A[(r&63)*a_sb + (r>>6)*a_ss + k].
// One block = 8 waves = one 16-row M tile x all 8 N tiles (one per wave).
// ---------------------------------------------------------------------------
__global__ __launch_bounds__(256) void rnn_proj_kernel(
    const float* __restrict__ A, const float* __restrict__ Wih,
    const float* __restrict__ bih, const float* __restrict__ bhh,
    float* __restrict__ P, int a_sb, int a_ss) {
    const int lane = threadIdx.x & 31;
    const int nt   = threadIdx.x >> 5;     // wave id == N tile 0..7
    const int mt   = blockIdx.x;           // M tile (16 rows of RB)

    const int col  = nt * 16 + (lane & 15);
    const float bias = bih[col] + bhh[col];

    const int r = mt * 16 + (lane & 15);
    const float* arow = A + (long)(r & 63) * a_sb + (long)(r >> 6) * a_ss;
    const int kbA = (lane >> 4) << 3;      // 0 or 8

    v8f acc = {};
#pragma unroll
    for (int kc = 0; kc < 4; ++kc) {
        const int kb = kc * 32 + kbA;
        v4f a0 = *(const v4f*)(arow + kb);
        v4f a1 = *(const v4f*)(arow + kb + 4);
        v4f a2 = *(const v4f*)(arow + kb + 16);
        v4f a3 = *(const v4f*)(arow + kb + 20);
        v16h af;
#pragma unroll
        for (int i = 0; i < 4; ++i) {
            af[i]      = (_Float16)a0[i];
            af[i + 4]  = (_Float16)a1[i];
            af[i + 8]  = (_Float16)a2[i];
            af[i + 12] = (_Float16)a3[i];
        }
        v16h bf = load_w_frag(Wih, kc, nt, lane);
        acc = __builtin_amdgcn_wmma_f32_16x16x32_f16(
            false, af, false, bf, (short)0, acc, false, false);
    }

#pragma unroll
    for (int v = 0; v < 8; ++v) {
        int row = mt * 16 + v + kbA;       // kbA == 8*(lane>=16)
        P[(long)row * HID + col] = acc[v] + bias;
    }
}

// ---------------------------------------------------------------------------
// Recurrent scan: persistent single workgroup (32 waves), full 64x128 h-update
// per timestep:  h = tanh(P[s] + h @ Whh^T).
//  * Whh fragments register-resident for all 2048 steps
//  * h double-buffered f16 in padded LDS rows
//  * P[s] tile (32KB) double-buffered in LDS, prefetched one step ahead with
//    global_load_async_to_lds_b128 (ASYNCcnt), waited with s_wait_asynccnt
//  * one __syncthreads per step
// Y index = row*y_rs + s*y_ss + col  (emits [S,B,H] or [B,S,H]).
// ---------------------------------------------------------------------------
__global__ __launch_bounds__(1024) void rnn_scan_kernel(
    const float* __restrict__ P, const float* __restrict__ Whh,
    float* __restrict__ Y, long y_rs, long y_ss) {
    extern __shared__ char smem[];
    float*    pbuf = (float*)smem;                       // [2][BATCH*HID]
    _Float16* hbuf = (_Float16*)(smem + PBUF_BYTES);     // [2][BATCH][HPAD]

    const int tid  = threadIdx.x;
    const int lane = tid & 31;
    const int wid  = tid >> 5;             // 0..31
    const int mt   = wid >> 3;             // 0..3  (batch tile)
    const int nt   = wid & 7;              // 0..7  (hidden tile)

    // Whh B fragments: resident in registers across all 2048 steps.
    v16h wb[4];
#pragma unroll
    for (int kc = 0; kc < 4; ++kc) wb[kc] = load_w_frag(Whh, kc, nt, lane);

    // h0 = 0
    for (int i = tid; i < 2 * BATCH * HPAD; i += 1024)
        hbuf[i] = (_Float16)0.0f;

    // async prefetch of P[0] into pbuf[0]: 1024 threads x 2 x 16B = 32KB
    {
        unsigned l0 = lds_off(pbuf) + (unsigned)tid * 16u;
        const float* g0 = P + tid * 4;
        asm volatile(
            "global_load_async_to_lds_b128 %0, %1, off\n\t"
            "global_load_async_to_lds_b128 %0, %1, off offset:16384"
            :: "v"(l0), "v"(g0) : "memory");
    }
    asm volatile("s_wait_asynccnt 0" ::: "memory");
    __syncthreads();

    const int col  = nt * 16 + (lane & 15);
    const int rowA = mt * 16 + (lane & 15);
    const int kbA  = (lane >> 4) << 3;     // 0 or 8

    for (int s = 0; s < SEQ; ++s) {
        const int cur = s & 1, nxt = cur ^ 1;
        const _Float16* hcur = hbuf + cur * BATCH * HPAD;
        _Float16*       hnxt = hbuf + nxt * BATCH * HPAD;
        const float*    pcur = pbuf + cur * BATCH * HID;

        // kick off async prefetch of P[s+1] into the other LDS buffer
        if (s + 1 < SEQ) {
            unsigned l0 = lds_off(pbuf + nxt * BATCH * HID) + (unsigned)tid * 16u;
            const float* g0 = P + (long)(s + 1) * (BATCH * HID) + tid * 4;
            asm volatile(
                "global_load_async_to_lds_b128 %0, %1, off\n\t"
                "global_load_async_to_lds_b128 %0, %1, off offset:16384"
                :: "v"(l0), "v"(g0) : "memory");
        }

        // acc starts at the prefetched input projection (bias already folded)
        v8f acc;
#pragma unroll
        for (int v = 0; v < 8; ++v)
            acc[v] = pcur[(mt * 16 + v + kbA) * HID + col];

        // 4 K-chunks of WMMA against h(t-1) held in LDS
#pragma unroll
        for (int kc = 0; kc < 4; ++kc) {
            const int kb = kc * 32 + kbA;
            v8h lo = *(const v8h*)&hcur[rowA * HPAD + kb];
            v8h hi = *(const v8h*)&hcur[rowA * HPAD + kb + 16];
            v16h af;
#pragma unroll
            for (int i = 0; i < 8; ++i) { af[i] = lo[i]; af[i + 8] = hi[i]; }
            acc = __builtin_amdgcn_wmma_f32_16x16x32_f16(
                false, af, false, wb[kc], (short)0, acc, false, false);
        }

        // tanh, emit f32 to global, write f16 h(t) into the other LDS buffer
#pragma unroll
        for (int v = 0; v < 8; ++v) {
            int row = mt * 16 + v + kbA;
            float t = tanhf(acc[v]);
            Y[(long)row * y_rs + (long)s * y_ss + col] = t;
            hnxt[row * HPAD + col] = (_Float16)t;
        }

        asm volatile("s_wait_asynccnt 0" ::: "memory");
        __syncthreads();   // one barrier per timestep
    }
}

extern "C" void kernel_launch(void* const* d_in, const int* in_sizes, int n_in,
                              void* d_out, int out_size, void* d_ws, size_t ws_size,
                              hipStream_t stream) {
    (void)in_sizes; (void)n_in; (void)out_size; (void)ws_size;
    const float* x    = (const float*)d_in[0];   // [B,S,H]
    const float* W_ih = (const float*)d_in[1];   // [2,H,H]
    const float* W_hh = (const float*)d_in[2];   // [2,H,H]
    const float* b_ih = (const float*)d_in[3];   // [2,H]
    const float* b_hh = (const float*)d_in[4];   // [2,H]
    float* out = (float*)d_out;                  // [B,S,H]

    float* P  = (float*)d_ws;                    // [S*B, H] projection buffer
    float* Y0 = (float*)d_ws + (size_t)RB * HID; // [S,B,H] layer-0 output

    const int mtiles = RB / 16;                  // 8192 blocks

    // ---- layer 0 ----
    rnn_proj_kernel<<<mtiles, 256, 0, stream>>>(
        x, W_ih, b_ih, b_hh, P, SEQ * HID, HID);
    rnn_scan_kernel<<<1, 1024, SMEM_BYTES, stream>>>(
        P, W_hh, Y0, (long)HID, (long)BATCH * HID);

    // ---- layer 1 ----
    rnn_proj_kernel<<<mtiles, 256, 0, stream>>>(
        Y0, W_ih + HID * HID, b_ih + HID, b_hh + HID, P, HID, BATCH * HID);
    rnn_scan_kernel<<<1, 1024, SMEM_BYTES, stream>>>(
        P, W_hh + HID * HID, out, (long)SEQ * HID, (long)HID);
}
